// MCO_25091198943739
// MI455X (gfx1250) — matrix operand order repaired and checked by reading, not compiled, not run on hardware
//
#include <hip/hip_runtime.h>
#include <hip/hip_bf16.h>

// ---------------------------------------------------------------------------
// MI455X (gfx1250) implementation.
// Heavy matmuls (conv1x1, conv3-as-GEMM K=768, attention projections, proj2)
// run through one WMMA bf16 GEMM kernel (v_wmma_f32_16x16x32_bf16, fp32
// accumulation), with double-buffered LDS tiles fed by the CDNA5 async
// load-to-LDS path (global_load_async_to_lds_b128 / s_wait_asynccnt).
// ---------------------------------------------------------------------------

#define USE_ASYNC_LDS 1   // gfx1250 async-to-LDS staging (set 0 to fall back)

typedef __attribute__((ext_vector_type(16))) __bf16 v16bf;
typedef __attribute__((ext_vector_type(8)))  __bf16 v8bf;
typedef __attribute__((ext_vector_type(8)))  float  v8f;
typedef __attribute__((ext_vector_type(4)))  float  v4f;

__device__ __forceinline__ __bf16 f2bf(float f) {
    unsigned u = __builtin_bit_cast(unsigned, f);
    unsigned r = (u + 0x7FFFu + ((u >> 16) & 1u)) >> 16;
    unsigned short s = (unsigned short)r;
    return __builtin_bit_cast(__bf16, s);
}
__device__ __forceinline__ float bf2f(__bf16 b) {
    unsigned short s = __builtin_bit_cast(unsigned short, b);
    unsigned u = ((unsigned)s) << 16;
    return __builtin_bit_cast(float, u);
}

// --- CDNA5 async global->LDS staging (CDNA5 ISA) ------
// LDS operand is the 32-bit LDS byte address (generic shared ptr truncated:
// low 32 bits kept, see ISA 10.2).
__device__ __forceinline__ unsigned lds_addr32(const void* p) {
    return (unsigned)(unsigned long long)(uintptr_t)p;
}
__device__ __forceinline__ void async_ld16(unsigned lds_off, const void* gaddr) {
#if USE_ASYNC_LDS
    asm volatile("global_load_async_to_lds_b128 %0, %1, off"
                 :: "v"(lds_off), "v"((unsigned long long)(uintptr_t)gaddr)
                 : "memory");
#else
    (void)lds_off; (void)gaddr;
#endif
}
__device__ __forceinline__ void wait_async0() {
#if USE_ASYNC_LDS
    asm volatile("s_wait_asynccnt 0" ::: "memory");
#endif
}

// ---------------------------------------------------------------------------
// Core WMMA GEMM: Y[row, m] = sum_k W[m, k] * X[row, k] (+ bias[m])
// W: [M, KC] bf16 row-major (M = gridDim.x*64).
// X rows addressed as X + (row / rows_per_b)*xbs + (row % rows_per_b)*ldx
// (same mapping for Y) -> supports flat tensors AND overlapping conv3
// windows over zero-padded [b, T+2, 256] buffers (ldx=256, KC=768).
// Block: 128 threads (4 waves). Tile: 64(M) x 64(N), K-step 32,
// double-buffered LDS ping-pong, one barrier per K-step.
// ---------------------------------------------------------------------------
template <bool OUT_BF16>
__global__ __launch_bounds__(128) void k_gemm_wmma(
    const __bf16* __restrict__ W, int KC,
    const __bf16* __restrict__ X,
    int rows_per_b, long long xbs, int ldx,
    const float* __restrict__ bias,
    void* __restrict__ Yv, long long ybs, int ldy)
{
    __shared__ __bf16 sW[2][64][48];   // 48-stride: 32B-aligned rows, bank spread
    __shared__ __bf16 sX[2][64][48];   // transposed tile: [n][k] (k contiguous)

    const int tid  = threadIdx.x;
    const int lane = tid & 31;
    const int wave = tid >> 5;
    const int m0 = blockIdx.x * 64;
    const int n0 = blockIdx.y * 64;

    // ---- staging addresses are loop-invariant: hoist everything ----
    const int r0 = tid >> 2,         co0 = (tid & 3) * 8;        // chunk 0
    const int r1 = (tid + 128) >> 2, co1 = ((tid + 128) & 3) * 8;// chunk 1
    const __bf16* wg0 = W + (long long)(m0 + r0) * KC + co0;
    const __bf16* wg1 = W + (long long)(m0 + r1) * KC + co1;
    auto rowptr = [&](int row) {
        return X + (long long)(row / rows_per_b) * xbs +
                   (long long)(row % rows_per_b) * ldx;
    };
    const __bf16* xg0 = rowptr(n0 + r0) + co0;
    const __bf16* xg1 = rowptr(n0 + r1) + co1;
    unsigned lw0[2], lw1[2], lx0[2], lx1[2];
#pragma unroll
    for (int bf = 0; bf < 2; ++bf) {
        lw0[bf] = lds_addr32(&sW[bf][r0][co0]);
        lw1[bf] = lds_addr32(&sW[bf][r1][co1]);
        lx0[bf] = lds_addr32(&sX[bf][r0][co0]);
        lx1[bf] = lds_addr32(&sX[bf][r1][co1]);
    }

#if !USE_ASYNC_LDS
    auto stage = [&](int bf, int k0) {
        *(v8bf*)(&sW[bf][r0][co0]) = *(const v8bf*)(wg0 + k0);
        *(v8bf*)(&sW[bf][r1][co1]) = *(const v8bf*)(wg1 + k0);
        *(v8bf*)(&sX[bf][r0][co0]) = *(const v8bf*)(xg0 + k0);
        *(v8bf*)(&sX[bf][r1][co1]) = *(const v8bf*)(xg1 + k0);
    };
#else
    auto stage = [&](int bf, int k0) {
        async_ld16(lw0[bf], wg0 + k0);
        async_ld16(lw1[bf], wg1 + k0);
        async_ld16(lx0[bf], xg0 + k0);
        async_ld16(lx1[bf], xg1 + k0);
    };
#endif

    v8f acc[4] = {};
    const int mrow  = wave * 16 + (lane & 15);
    const int abase = (lane >> 4) * 8;
    const int klo   = (lane >> 4) * 16;
    const int nkit  = KC >> 5;

    stage(0, 0);                 // preload tile 0
    wait_async0();
    __syncthreads();

    for (int it = 0; it < nkit; ++it) {
        const int buf = it & 1;
        if (it + 1 < nkit) stage(buf ^ 1, (it + 1) * 32);  // overlap with WMMA

        // A fragment (16x32): lane L -> m = 16w + (L&15); k = (L>>4)*16 .. +15
        v16bf afrag = *(const v16bf*)(&sW[buf][mrow][klo]);
        // B fragment (32x16): lane L -> n = (L&15); k = (L>>4)*16 .. +15
#pragma unroll
        for (int j = 0; j < 4; ++j) {
            const int nrow = j * 16 + (lane & 15);
            v16bf bfrag = *(const v16bf*)(&sX[buf][nrow][klo]);
            acc[j] = __builtin_amdgcn_wmma_f32_16x16x32_bf16(
                false, afrag, false, bfrag, (short)0, acc[j], false, false);
        }
        wait_async0();           // next tile landed in LDS
        __syncthreads();         // everyone done reading buf before reuse
    }

    // Epilogue: per lane, acc[j] elems r=0..7 are consecutive m values.
    const int hi4   = lane >> 4;
    const int mbase = m0 + wave * 16 + hi4 * 8;
    float badd[8];
#pragma unroll
    for (int r = 0; r < 8; ++r) badd[r] = bias ? bias[mbase + r] : 0.0f;

#pragma unroll
    for (int j = 0; j < 4; ++j) {
        const int row = n0 + j * 16 + (lane & 15);
        long long yoff = (long long)(row / rows_per_b) * ybs +
                         (long long)(row % rows_per_b) * ldy + mbase;
        if (OUT_BF16) {
            __bf16* Y = (__bf16*)Yv;
            v8bf o;
#pragma unroll
            for (int r = 0; r < 8; ++r) o[r] = f2bf(acc[j][r] + badd[r]);
            *(v8bf*)(Y + yoff) = o;
        } else {
            float* Y = (float*)Yv;
            v4f o0, o1;
#pragma unroll
            for (int r = 0; r < 4; ++r) { o0[r] = acc[j][r] + badd[r];
                                          o1[r] = acc[j][r + 4] + badd[r + 4]; }
            *(v4f*)(Y + yoff)     = o0;
            *(v4f*)(Y + yoff + 4) = o1;
        }
    }
}

// ---------------------------------------------------------------------------
// Layout / conversion helpers
// ---------------------------------------------------------------------------
__global__ void k_cvt(const float* __restrict__ X, __bf16* __restrict__ Y, long long n) {
    long long i = (long long)blockIdx.x * blockDim.x + threadIdx.x;
    if (i < n) Y[i] = f2bf(X[i]);
}

__global__ void k_fill_bf16(__bf16* __restrict__ p, long long n) {
    long long i = (long long)blockIdx.x * blockDim.x + threadIdx.x;
    if (i < n) p[i] = f2bf(0.0f);
}

// fp32 [B,C,T] -> bf16 [B,T,ldo] at column offset co ("nk" layout)
__global__ void k_transpose_cvt(const float* __restrict__ X, __bf16* __restrict__ Y,
                                int C, int T, int ldo, int co) {
    __shared__ float tile[32][33];
    int b = blockIdx.z, t0 = blockIdx.x * 32, c0 = blockIdx.y * 32;
    int tx = threadIdx.x, ty = threadIdx.y;
    for (int i = ty; i < 32; i += 8)
        tile[i][tx] = X[((long long)b * C + c0 + i) * T + t0 + tx];
    __syncthreads();
    for (int i = ty; i < 32; i += 8)
        Y[((long long)b * T + t0 + i) * ldo + co + c0 + tx] = f2bf(tile[tx][i]);
}

// ffn weight [O=256][I=256][3] -> bf16 [O][tau*256 + i] (tau-major reduction)
__global__ void k_reorder_w3(const float* __restrict__ W, __bf16* __restrict__ Y) {
    int idx = blockIdx.x * 256 + threadIdx.x;  // 256*768
    int o = idx / 768, r = idx % 768, tau = r >> 8, i = r & 255;
    Y[idx] = f2bf(W[((long long)o * 256 + i) * 3 + tau]);
}

// ---------------------------------------------------------------------------
// BatchNorm (training-mode, stats over all rows) — deterministic 2-stage
// ---------------------------------------------------------------------------
__global__ void k_colstats_partial(const float* __restrict__ X, int rows_per_block,
                                   float* __restrict__ part) {
    int c = threadIdx.x, b = blockIdx.x;
    long long r0 = (long long)b * rows_per_block;
    float s = 0.f, s2 = 0.f;
    for (int r = 0; r < rows_per_block; ++r) {
        float v = X[(r0 + r) * 256 + c];
        s += v; s2 += v * v;
    }
    part[((long long)b * 2 + 0) * 256 + c] = s;
    part[((long long)b * 2 + 1) * 256 + c] = s2;
}

__global__ void k_colstats_final(const float* __restrict__ part, int nblocks, float invN,
                                 float* __restrict__ mean, float* __restrict__ rstd) {
    int c = threadIdx.x;
    float s = 0.f, s2 = 0.f;
    for (int b = 0; b < nblocks; ++b) {
        s  += part[((long long)b * 2 + 0) * 256 + c];
        s2 += part[((long long)b * 2 + 1) * 256 + c];
    }
    float m = s * invN, v = s2 * invN - m * m;
    mean[c] = m;
    rstd[c] = rsqrtf(v + 1e-5f);
}

// normalize + ReLU -> bf16 (flat or padded destination via row mapping)
__global__ void k_bn_apply(const float* __restrict__ X, const float* __restrict__ mean,
                           const float* __restrict__ rstd, const float* __restrict__ g,
                           const float* __restrict__ be, __bf16* __restrict__ Y,
                           int rows_per_b, long long ybs, int ldy) {
    int row = blockIdx.x, c = threadIdx.x;
    float v = (X[(long long)row * 256 + c] - mean[c]) * rstd[c] * g[c] + be[c];
    v = fmaxf(v, 0.0f);
    long long yoff = (long long)(row / rows_per_b) * ybs +
                     (long long)(row % rows_per_b) * ldy + c;
    Y[yoff] = f2bf(v);
}

// ---------------------------------------------------------------------------
// branch logits (M=8 -> VALU dot), softmax over 3 branches + nf combine
// ---------------------------------------------------------------------------
__global__ void k_logits(const __bf16* __restrict__ H, const float* __restrict__ W2,
                         const float* __restrict__ b2, float* __restrict__ L) {
    int idx = blockIdx.x * blockDim.x + threadIdx.x;  // 16384*8
    int row = idx >> 3, k = idx & 7;
    const __bf16* h = H + (long long)row * 256;
    const float* w = W2 + k * 256;
    float a = 0.f;
    for (int i = 0; i < 256; ++i) a += bf2f(h[i]) * w[i];
    L[idx] = a + b2[k];
}

__global__ void k_nfcombine(const float* __restrict__ L,
                            const __bf16* __restrict__ f1nk,
                            const __bf16* __restrict__ f2nk,
                            const __bf16* __restrict__ f3nk,
                            __bf16* __restrict__ nfpad) {
    __shared__ float att[3][8];
    int row = blockIdx.x;          // (b,t): b = row/512, t = row%512
    int d = threadIdx.x;           // channel
    if (d < 8) {
        float l0 = L[(long long)row * 8 + d];
        float l1 = L[131072 + (long long)row * 8 + d];
        float l2 = L[262144 + (long long)row * 8 + d];
        float mx = fmaxf(l0, fmaxf(l1, l2));
        float e0 = __expf(l0 - mx), e1 = __expf(l1 - mx), e2 = __expf(l2 - mx);
        float inv = 1.0f / (e0 + e1 + e2);
        att[0][d] = e0 * inv; att[1][d] = e1 * inv; att[2][d] = e2 * inv;
    }
    __syncthreads();
    int b = row >> 9, t = row & 511;
    float v0 = bf2f(f1nk[(long long)row * 256 + d]);
    float v1 = bf2f(f2nk[(long long)row * 256 + d]);
    float v2 = bf2f(f3nk[(long long)row * 256 + d]);
    long long base = ((long long)b * 514 + t + 1) * 256 + d;  // +1 = pad row
    const long long kstride = 32LL * 514 * 256;
#pragma unroll
    for (int k = 0; k < 8; ++k) {
        float o = att[0][k] * v0 + att[1][k] * v1 + att[2][k] * v2;
        nfpad[k * kstride + base] = f2bf(o);
    }
}

// BN2 + ReLU + residual(nf) + avgpool2 -> kv rows [b*T2, 8, 256]
__global__ void k_bn_res_pool(const float* __restrict__ X2, const float* __restrict__ mean,
                              const float* __restrict__ rstd, const float* __restrict__ g,
                              const float* __restrict__ be, const __bf16* __restrict__ nfk,
                              __bf16* __restrict__ kv, int kbranch) {
    int idx = blockIdx.x * blockDim.x + threadIdx.x;  // 8192*256
    int c = idx & 255, r2 = idx >> 8;
    int b = r2 >> 8, t2 = r2 & 255, t = t2 * 2;
    float m = mean[c], rs = rstd[c], gg = g[c], bb = be[c];
    long long rowA = (long long)b * 512 + t;
    float x0 = fmaxf((X2[rowA * 256 + c] - m) * rs * gg + bb, 0.f) +
               bf2f(nfk[((long long)b * 514 + t + 1) * 256 + c]);
    float x1 = fmaxf((X2[(rowA + 1) * 256 + c] - m) * rs * gg + bb, 0.f) +
               bf2f(nfk[((long long)b * 514 + t + 2) * 256 + c]);
    kv[((long long)r2 * 8 + kbranch) * 256 + c] = f2bf(0.5f * (x0 + x1));
}

// query: avgpool2 over t then negate -> bf16 rows [8192, 256]
__global__ void k_qpool(const float* __restrict__ Xq, __bf16* __restrict__ qrow) {
    int idx = blockIdx.x * blockDim.x + threadIdx.x;  // 8192*256
    int c = idx & 255, r2 = idx >> 8;
    int b = r2 >> 8, t2 = r2 & 255;
    long long rr = ((long long)b * 512 + 2 * t2) * 256 + c;
    qrow[idx] = f2bf(-0.5f * (Xq[rr] + Xq[rr + 256]));
}

// policy 1x1 conv over cat(f1,f2,f3): M=8 -> VALU dot
__global__ void k_policy(const __bf16* __restrict__ f1nk, const __bf16* __restrict__ f2nk,
                         const __bf16* __restrict__ f3nk, const float* __restrict__ Wp,
                         const float* __restrict__ bp, float* __restrict__ pl) {
    int idx = blockIdx.x * blockDim.x + threadIdx.x;  // 16384*8
    int row = idx >> 3, k = idx & 7;
    const float* w = Wp + k * 768;
    const __bf16* a0 = f1nk + (long long)row * 256;
    const __bf16* a1 = f2nk + (long long)row * 256;
    const __bf16* a2 = f3nk + (long long)row * 256;
    float s = bp[k];
    for (int i = 0; i < 256; ++i)
        s += bf2f(a0[i]) * w[i] + bf2f(a1[i]) * w[256 + i] + bf2f(a2[i]) * w[512 + i];
    pl[idx] = s;
}

// gumbel-softmax hard argmax -> ip (to d_out) + threshold mask
__global__ void k_gumbel(const float* __restrict__ pl, const float* __restrict__ gn,
                         const float* __restrict__ tau, float* __restrict__ ip_out,
                         float* __restrict__ mask) {
    int r2 = blockIdx.x * blockDim.x + threadIdx.x;  // 8192
    if (r2 >= 8192) return;
    int b = r2 >> 8, t2 = r2 & 255, t = 2 * t2;
    float itau = 1.0f / tau[0];
    float ymax = -3.4e38f; int amax = 0;
#pragma unroll
    for (int k = 0; k < 8; ++k) {
        float p = 0.5f * (pl[((long long)(b * 512 + t)) * 8 + k] +
                          pl[((long long)(b * 512 + t + 1)) * 8 + k]);
        float y = (p + gn[((long long)b * 8 + k) * 256 + t2]) * itau;
        if (y > ymax) { ymax = y; amax = k; }
    }
#pragma unroll
    for (int k = 0; k < 8; ++k) {
        ip_out[((long long)b * 8 + k) * 256 + t2] = (k == amax) ? 1.0f : 0.0f;
        mask[(long long)r2 * 8 + k] = (k > amax) ? -1e9f : 0.0f;
    }
}

// final masked MHA: q_len=1, kv_len=8, H=4 heads of 64 -> out [b, d, t2]
__global__ __launch_bounds__(256) void k_attn(
    const __bf16* __restrict__ Qp, const __bf16* __restrict__ Kp,
    const __bf16* __restrict__ Vp, const float* __restrict__ mask,
    float* __restrict__ out) {
    __shared__ float prod[8][256];
    __shared__ float scs[4][8];
    __shared__ float attw[4][8];
    int r = blockIdx.x, d = threadIdx.x, h = d >> 6;
    float q = bf2f(Qp[(long long)r * 256 + d]);
#pragma unroll
    for (int k = 0; k < 8; ++k)
        prod[k][d] = q * bf2f(Kp[((long long)r * 8 + k) * 256 + d]);
    __syncthreads();
    if (d < 32) {
        int k = d >> 2, hh = d & 3;
        float s = 0.f;
        for (int j = 0; j < 64; ++j) s += prod[k][hh * 64 + j];
        scs[hh][k] = -(s * 0.125f) + mask[(long long)r * 8 + k];
    }
    __syncthreads();
    if (d < 4) {
        float mx = scs[d][0];
#pragma unroll
        for (int k = 1; k < 8; ++k) mx = fmaxf(mx, scs[d][k]);
        float e[8], sum = 0.f;
#pragma unroll
        for (int k = 0; k < 8; ++k) { e[k] = __expf(scs[d][k] - mx); sum += e[k]; }
        float inv = 1.0f / sum;
#pragma unroll
        for (int k = 0; k < 8; ++k) attw[d][k] = e[k] * inv;
    }
    __syncthreads();
    float o = 0.f;
#pragma unroll
    for (int k = 0; k < 8; ++k)
        o += attw[h][k] * bf2f(Vp[((long long)r * 8 + k) * 256 + d]);
    int b = r >> 8, t2 = r & 255;
    out[((long long)b * 256 + d) * 256 + t2] = o;
}

// ---------------------------------------------------------------------------
// Orchestration
// ---------------------------------------------------------------------------
extern "C" void kernel_launch(void* const* d_in, const int* in_sizes, int n_in,
                              void* d_out, int out_size, void* d_ws, size_t ws_size,
                              hipStream_t stream) {
    (void)in_sizes; (void)n_in; (void)out_size; (void)ws_size;
    const float* f1    = (const float*)d_in[0];
    const float* f2    = (const float*)d_in[1];
    const float* f3    = (const float*)d_in[2];
    const float* c_    = (const float*)d_in[3];
    const float* s_    = (const float*)d_in[4];
    const float* gn    = (const float*)d_in[5];
    const float* tau   = (const float*)d_in[6];
    const float* p1w1  = (const float*)d_in[7];
    const float* p1b1  = (const float*)d_in[8];
    const float* p1g1  = (const float*)d_in[9];
    const float* p1be1 = (const float*)d_in[10];
    const float* p1w2  = (const float*)d_in[11];
    const float* p1b2  = (const float*)d_in[12];
    const float* fw1   = (const float*)d_in[13];
    const float* fb1   = (const float*)d_in[14];
    const float* fg1   = (const float*)d_in[15];
    const float* fbe1  = (const float*)d_in[16];
    const float* fw2   = (const float*)d_in[17];
    const float* fb2   = (const float*)d_in[18];
    const float* fg2   = (const float*)d_in[19];
    const float* fbe2  = (const float*)d_in[20];
    const float* wq    = (const float*)d_in[21];
    const float* bq    = (const float*)d_in[22];
    const float* wk    = (const float*)d_in[23];
    const float* bk    = (const float*)d_in[24];
    const float* wv    = (const float*)d_in[25];
    const float* bv    = (const float*)d_in[26];
    const float* p2w   = (const float*)d_in[27];
    const float* p2b   = (const float*)d_in[28];
    const float* pow_  = (const float*)d_in[29];
    const float* pob   = (const float*)d_in[30];

    char* base = (char*)d_ws;
    size_t off = 0;
    auto alloc = [&](size_t bytes) -> void* {
        void* r = base + off;
        off += (bytes + 255) & ~(size_t)255;
        return r;
    };
    const long long PADR = 514, KSTR = 32LL * 514 * 256;

    __bf16* f1nk  = (__bf16*)alloc(16384LL * 256 * 2);
    __bf16* f2nk  = (__bf16*)alloc(16384LL * 256 * 2);
    __bf16* f3nk  = (__bf16*)alloc(16384LL * 256 * 2);
    __bf16* csnk  = (__bf16*)alloc(16384LL * 512 * 2);
    __bf16* hnk   = (__bf16*)alloc(16384LL * 256 * 2);
    __bf16* nfpad = (__bf16*)alloc(8LL * 32 * PADR * 256 * 2);
    __bf16* h1pad = (__bf16*)alloc(32LL * PADR * 256 * 2);
    __bf16* kvb   = (__bf16*)alloc(8192LL * 8 * 256 * 2);
    __bf16* qrow  = (__bf16*)alloc(8192LL * 256 * 2);
    __bf16* Qp    = (__bf16*)alloc(8192LL * 256 * 2);
    __bf16* Kpb   = (__bf16*)alloc(65536LL * 256 * 2);
    __bf16* Vpb   = (__bf16*)alloc(65536LL * 256 * 2);
    __bf16* p1w1b = (__bf16*)alloc(3LL * 65536 * 2);
    __bf16* wqb   = (__bf16*)alloc(65536LL * 2);
    __bf16* wkb   = (__bf16*)alloc(65536LL * 2);
    __bf16* wvb   = (__bf16*)alloc(65536LL * 2);
    __bf16* p2wb  = (__bf16*)alloc(131072LL * 2);
    __bf16* ffn1b = (__bf16*)alloc(8LL * 196608 * 2);
    __bf16* ffn2b = (__bf16*)alloc(8LL * 196608 * 2);
    float* hpre   = (float*)alloc(16384LL * 256 * 4);
    float* part   = (float*)alloc(64LL * 2 * 256 * 4);
    float* meanb  = (float*)alloc(256 * 4);
    float* rstdb  = (float*)alloc(256 * 4);
    float* Lbuf   = (float*)alloc(3LL * 131072 * 4);
    float* plbuf  = (float*)alloc(131072LL * 4);
    float* maskb  = (float*)alloc(8192LL * 8 * 4);

    const int FLAT = 1 << 30;  // rows_per_b for non-batched mappings

    // --- weight conversion / reorder ---
    k_cvt<<<768, 256, 0, stream>>>(p1w1, p1w1b, 3LL * 65536);
    k_cvt<<<256, 256, 0, stream>>>(wq, wqb, 65536);
    k_cvt<<<256, 256, 0, stream>>>(wk, wkb, 65536);
    k_cvt<<<256, 256, 0, stream>>>(wv, wvb, 65536);
    k_cvt<<<512, 256, 0, stream>>>(p2w, p2wb, 131072);
    for (int kb = 0; kb < 8; ++kb) {
        k_reorder_w3<<<768, 256, 0, stream>>>(fw1 + (long long)kb * 256 * 256 * 3,
                                              ffn1b + (long long)kb * 196608);
        k_reorder_w3<<<768, 256, 0, stream>>>(fw2 + (long long)kb * 256 * 256 * 3,
                                              ffn2b + (long long)kb * 196608);
    }

    // --- activation transposes to nk layout; zero padded buffers ---
    dim3 tb(32, 8);
    k_transpose_cvt<<<dim3(16, 8, 32), tb, 0, stream>>>(f1, f1nk, 256, 512, 256, 0);
    k_transpose_cvt<<<dim3(16, 8, 32), tb, 0, stream>>>(f2, f2nk, 256, 512, 256, 0);
    k_transpose_cvt<<<dim3(16, 8, 32), tb, 0, stream>>>(f3, f3nk, 256, 512, 256, 0);
    k_transpose_cvt<<<dim3(16, 8, 32), tb, 0, stream>>>(c_, csnk, 256, 512, 512, 0);
    k_transpose_cvt<<<dim3(16, 8, 32), tb, 0, stream>>>(s_, csnk, 256, 512, 512, 256);
    {
        long long n1 = 8LL * 32 * PADR * 256;
        k_fill_bf16<<<(int)((n1 + 255) / 256), 256, 0, stream>>>(nfpad, n1);
        long long n2 = 32LL * PADR * 256;
        k_fill_bf16<<<(int)((n2 + 255) / 256), 256, 0, stream>>>(h1pad, n2);
    }

    // --- branch logits: conv1x1 + BN + ReLU + conv(->K) per branch ---
    for (int m = 0; m < 3; ++m) {
        const __bf16* fnk = (m == 0) ? f1nk : (m == 1 ? f2nk : f3nk);
        k_gemm_wmma<false><<<dim3(4, 256), 128, 0, stream>>>(
            p1w1b + (long long)m * 65536, 256, fnk, FLAT, 0, 256,
            p1b1 + m * 256, hpre, 0, 256);
        k_colstats_partial<<<64, 256, 0, stream>>>(hpre, 256, part);
        k_colstats_final<<<1, 256, 0, stream>>>(part, 64, 1.0f / 16384.0f, meanb, rstdb);
        k_bn_apply<<<16384, 256, 0, stream>>>(hpre, meanb, rstdb, p1g1 + m * 256,
                                              p1be1 + m * 256, hnk, FLAT, 0, 256);
        k_logits<<<512, 256, 0, stream>>>(hnk, p1w2 + m * 8 * 256, p1b2 + m * 8,
                                          Lbuf + (long long)m * 131072);
    }

    // --- softmax over branches + weighted combine into padded nf buffers ---
    k_nfcombine<<<16384, 256, 0, stream>>>(Lbuf, f1nk, f2nk, f3nk, nfpad);

    // --- FFN: (conv3+BN+ReLU) x2 + residual + avgpool2, per K branch ---
    for (int kb = 0; kb < 8; ++kb) {
        const __bf16* nfk = nfpad + kb * KSTR;
        k_gemm_wmma<false><<<dim3(4, 256), 128, 0, stream>>>(
            ffn1b + (long long)kb * 196608, 768, nfk, 512, PADR * 256, 256,
            fb1 + kb * 256, hpre, 0, 256);
        k_colstats_partial<<<64, 256, 0, stream>>>(hpre, 256, part);
        k_colstats_final<<<1, 256, 0, stream>>>(part, 64, 1.0f / 16384.0f, meanb, rstdb);
        k_bn_apply<<<16384, 256, 0, stream>>>(hpre, meanb, rstdb, fg1 + kb * 256,
                                              fbe1 + kb * 256, h1pad + 256,
                                              512, PADR * 256, 256);
        k_gemm_wmma<false><<<dim3(4, 256), 128, 0, stream>>>(
            ffn2b + (long long)kb * 196608, 768, h1pad, 512, PADR * 256, 256,
            fb2 + kb * 256, hpre, 0, 256);
        k_colstats_partial<<<64, 256, 0, stream>>>(hpre, 256, part);
        k_colstats_final<<<1, 256, 0, stream>>>(part, 64, 1.0f / 16384.0f, meanb, rstdb);
        k_bn_res_pool<<<8192, 256, 0, stream>>>(hpre, meanb, rstdb, fg2 + kb * 256,
                                                fbe2 + kb * 256, nfk, kvb, kb);
    }

    // --- query path: proj2 GEMM (K=512) + pool + negate ---
    k_gemm_wmma<false><<<dim3(4, 256), 128, 0, stream>>>(
        p2wb, 512, csnk, FLAT, 0, 512, p2b, hpre, 0, 256);
    k_qpool<<<8192, 256, 0, stream>>>(hpre, qrow);

    // --- policy + gumbel hard mask (ip written straight to output) ---
    k_policy<<<512, 256, 0, stream>>>(f1nk, f2nk, f3nk, pow_, pob, plbuf);
    k_gumbel<<<32, 256, 0, stream>>>(plbuf, gn, tau,
                                     (float*)d_out + 32LL * 256 * 256, maskb);

    // --- attention projections (WMMA GEMMs) ---
    k_gemm_wmma<true><<<dim3(4, 128), 128, 0, stream>>>(
        wqb, 256, qrow, FLAT, 0, 256, bq, Qp, 0, 256);
    k_gemm_wmma<true><<<dim3(4, 1024), 128, 0, stream>>>(
        wkb, 256, kvb, FLAT, 0, 256, bk, Kpb, 0, 256);
    k_gemm_wmma<true><<<dim3(4, 1024), 128, 0, stream>>>(
        wvb, 256, kvb, FLAT, 0, 256, bv, Vpb, 0, 256);

    // --- final masked MHA -> out [b, d, t2] ---
    k_attn<<<8192, 256, 0, stream>>>(Qp, Kpb, Vpb, maskb, (float*)d_out);
}
